// CrossDepthAttention_58677843198035
// MI455X (gfx1250) — matrix operand order repaired and checked by reading, not compiled, not run on hardware
//
#include <hip/hip_runtime.h>
#include <cstdint>

// ---------------------------------------------------------------------------
// CrossDepthAttention for MI455X (gfx1250, wave32, WMMA).
// ~309 GFLOP of K=1024 GEMMs (compute-bound at bf16-WMMA rates) + tiny softmax.
// Pipeline:
//   0) bulk f32->bf16 conversion of current/history/weights (native cvt)
//   1) Q = cur @ Wq^T                  (WMMA, async-LDS double buffered)
//   2) K,V = hist @ {Wk,Wv}^T fused    (WMMA, shares history reads)
//   3) per-(token,head) softmax over depth N=8 (wave32 + shfl_xor)
//   4) out = cur + attn @ Wo^T  (fp32) (WMMA + residual)
// ---------------------------------------------------------------------------

typedef __attribute__((ext_vector_type(16))) __bf16 v16bf;
typedef __attribute__((ext_vector_type(8)))  float  v8f;

__device__ __forceinline__ unsigned short f32_to_bf16(float f) {
  union { __bf16 h; unsigned short u; } c;
  c.h = (__bf16)f;               // native RNE convert
  return c.u;
}
__device__ __forceinline__ unsigned pk_bf16(float a, float b) {
  return (unsigned)f32_to_bf16(a) | ((unsigned)f32_to_bf16(b) << 16);
}
__device__ __forceinline__ float bf16_to_f32(unsigned short h) {
  union { unsigned u; float f; } x; x.u = ((unsigned)h) << 16;
  return x.f;
}

// --- CDNA5 async copy: global -> LDS, 16B per lane, ASYNCcnt tracked -------
__device__ __forceinline__ void async_b128(unsigned lds_off,
                                           const unsigned short* g) {
  asm volatile("global_load_async_to_lds_b128 %0, %1, off"
               :: "v"(lds_off), "v"((unsigned long long)(uintptr_t)g)
               : "memory");
}
template <int N> __device__ __forceinline__ void wait_asynccnt() {
#if __has_builtin(__builtin_amdgcn_s_wait_asynccnt)
  __builtin_amdgcn_s_wait_asynccnt(N);
#else
  asm volatile("s_wait_asynccnt %0" :: "n"(N) : "memory");
#endif
}

// ---------------------------------------------------------------------------
// Bulk f32 -> bf16 (8 elements / thread, exact grids, all sizes % 2048 == 0)
// ---------------------------------------------------------------------------
__global__ void __launch_bounds__(256)
cvt_bf16_kernel(const float* __restrict__ src, unsigned short* __restrict__ dst) {
  const long long i = (long long)blockIdx.x * 256 + threadIdx.x;
  const float4 a = ((const float4*)src)[2 * i];
  const float4 b = ((const float4*)src)[2 * i + 1];
  uint4 o;
  o.x = pk_bf16(a.x, a.y);
  o.y = pk_bf16(a.z, a.w);
  o.z = pk_bf16(b.x, b.y);
  o.w = pk_bf16(b.z, b.w);
  ((uint4*)dst)[i] = o;
}

// ---------------------------------------------------------------------------
// All-bf16 WMMA GEMM:  C[o] = A(Mx1024) @ W[o](1024x1024)^T
//   Tile 128x64x32, 256 threads = 8 waves; wave owns 32x32 per output.
//   Double-buffered LDS fed by global_load_async_to_lds_b128:
//     issue tile k+1, s_wait_asynccnt <= P (tile k done, k+1 in flight),
//     barrier, ds_load fragments, 4*NOUT wmma, barrier.
//   LDS row stride 48 bf16 (96B) keeps every 16B chunk aligned.
// ---------------------------------------------------------------------------
#define LDSW 48

template <int NOUT, bool RESIDUAL>
__global__ void __launch_bounds__(256)
gemm_wmma_async(const unsigned short* __restrict__ A,
                const unsigned short* __restrict__ W0,
                const unsigned short* __restrict__ W1,
                void* __restrict__ C0v, void* __restrict__ C1v,
                const float* __restrict__ resid) {
  __shared__ unsigned short Alds[2][128 * LDSW];
  __shared__ unsigned short Wlds[NOUT][2][64 * LDSW];

  constexpr int K = 1024, Ndim = 1024, NK = K / 32;
  const int tid  = threadIdx.x;
  const int lane = tid & 31;
  const int wave = tid >> 5;
  const int bm = blockIdx.y * 128;
  const int bn = blockIdx.x * 64;
  const int mw = (wave >> 1) * 32;  // 0,32,64,96
  const int nw = (wave & 1) * 32;   // 0,32
  const int half = lane >> 4;
  const int r15  = lane & 15;

  // ---- per-thread staging geometry: one 16B chunk id per async instr ----
  const int ar = tid >> 2;            // A rows 0..63 (+64 for 2nd instr)
  const int ce = (tid & 3) * 8;       // chunk column, in bf16 elements
  const unsigned short* gA0 = A + (size_t)(bm + ar) * K + ce;
  const unsigned short* gA1 = A + (size_t)(bm + 64 + ar) * K + ce;
  const unsigned short* Wsrc[2] = {W0, W1};
  const unsigned short* gW[NOUT];
#pragma unroll
  for (int o = 0; o < NOUT; ++o) gW[o] = Wsrc[o] + (size_t)(bn + ar) * K + ce;

  unsigned aoff0[2], aoff1[2], woff[NOUT][2];
#pragma unroll
  for (int b = 0; b < 2; ++b) {
    aoff0[b] = (unsigned)(uintptr_t)&Alds[b][ar * LDSW + ce];
    aoff1[b] = (unsigned)(uintptr_t)&Alds[b][(64 + ar) * LDSW + ce];
#pragma unroll
    for (int o = 0; o < NOUT; ++o)
      woff[o][b] = (unsigned)(uintptr_t)&Wlds[o][b][ar * LDSW + ce];
  }

  auto prefetch = [&](int kb, int b) {
    async_b128(aoff0[b], gA0 + kb);
    async_b128(aoff1[b], gA1 + kb);
#pragma unroll
    for (int o = 0; o < NOUT; ++o) async_b128(woff[o][b], gW[o] + kb);
  };

  v8f acc[NOUT][2][2];
#pragma unroll
  for (int o = 0; o < NOUT; ++o)
#pragma unroll
    for (int i = 0; i < 2; ++i)
#pragma unroll
      for (int j = 0; j < 2; ++j) acc[o][i][j] = (v8f){0,0,0,0,0,0,0,0};

  prefetch(0, 0);

#pragma unroll 2
  for (int k = 0; k < NK; ++k) {
    const int buf = k & 1;
    if (k + 1 < NK) {
      prefetch((k + 1) * 32, buf ^ 1);
      wait_asynccnt<NOUT + 2>();   // tile k complete; k+1 still in flight
    } else {
      wait_asynccnt<0>();
    }
    __syncthreads();               // tile k visible to all waves

    // A fragments (16x32): lanes 0-15 K{0..7,16..23}, lanes 16-31 K{8..15,24..31}
    v16bf afrag[2];
#pragma unroll
    for (int i = 0; i < 2; ++i) {
      const unsigned short* base =
          &Alds[buf][(mw + i * 16 + r15) * LDSW + half * 8];
      union { uint4 u[2]; v16bf v; } fa;
      fa.u[0] = *(const uint4*)(base);
      fa.u[1] = *(const uint4*)(base + 16);
      afrag[i] = fa.v;
    }
#pragma unroll
    for (int o = 0; o < NOUT; ++o) {
#pragma unroll
      for (int j = 0; j < 2; ++j) {
        // B fragment (32x16): lane n K{0..7,16..23}, lane n+16 K{8..15,24..31}; col n == W row n
        const unsigned short* base =
            &Wlds[o][buf][(nw + j * 16 + r15) * LDSW + half * 8];
        union { uint4 u[2]; v16bf v; } fb;
        fb.u[0] = *(const uint4*)(base);
        fb.u[1] = *(const uint4*)(base + 16);
        const v16bf bfrag = fb.v;
#pragma unroll
        for (int i = 0; i < 2; ++i) {
          acc[o][i][j] = __builtin_amdgcn_wmma_f32_16x16x32_bf16(
              false, afrag[i], false, bfrag, (short)0, acc[o][i][j],
              false, false);
        }
      }
    }
    __syncthreads();               // all reads of buf done before it is refilled
  }

  // ---- writeback: lane<16 -> M=e, N=lane; lane>=16 -> M=8+e, N=lane-16 ----
#pragma unroll
  for (int o = 0; o < NOUT; ++o) {
#pragma unroll
    for (int i = 0; i < 2; ++i) {
#pragma unroll
      for (int j = 0; j < 2; ++j) {
        const int n  = bn + nw + j * 16 + r15;
        const int m0 = bm + mw + i * 16 + half * 8;
#pragma unroll
        for (int e = 0; e < 8; ++e) {
          const size_t idx = (size_t)(m0 + e) * Ndim + n;
          if (RESIDUAL) {
            ((float*)C0v)[idx] = resid[idx] + acc[o][i][j][e];
          } else {
            unsigned short* dst = (unsigned short*)(o ? C1v : C0v);
            dst[idx] = f32_to_bf16(acc[o][i][j][e]);
          }
        }
      }
    }
  }
}

// ---------------------------------------------------------------------------
// Depth softmax: one wave per (token, head); lanes split Hd=64 (2 each).
// ---------------------------------------------------------------------------
__global__ void __launch_bounds__(256)
depth_softmax_kernel(const unsigned short* __restrict__ Q,
                     const unsigned short* __restrict__ Kb,
                     const unsigned short* __restrict__ Vb,
                     unsigned short* __restrict__ AO) {
  const int lane = threadIdx.x & 31;
  const int w = blockIdx.x * 8 + (threadIdx.x >> 5);
  const int t = w >> 4;
  const int h = w & 15;
  const size_t qbase = (size_t)t * 1024 + h * 64;

  const float q0 = bf16_to_f32(Q[qbase + lane]);
  const float q1 = bf16_to_f32(Q[qbase + lane + 32]);

  float logits[8];
  const float scale = 0.125f;  // 1/sqrt(64)
#pragma unroll
  for (int n = 0; n < 8; ++n) {
    const size_t kbase = ((size_t)t * 8 + n) * 1024 + h * 64;
    float p = q0 * bf16_to_f32(Kb[kbase + lane]) +
              q1 * bf16_to_f32(Kb[kbase + lane + 32]);
#pragma unroll
    for (int off = 16; off > 0; off >>= 1) p += __shfl_xor(p, off, 32);
    logits[n] = p * scale;
  }
  float mx = logits[0];
#pragma unroll
  for (int n = 1; n < 8; ++n) mx = fmaxf(mx, logits[n]);
  float e[8], denom = 0.f;
#pragma unroll
  for (int n = 0; n < 8; ++n) { e[n] = __expf(logits[n] - mx); denom += e[n]; }
  const float inv = __frcp_rn(denom);

  float o0 = 0.f, o1 = 0.f;
#pragma unroll
  for (int n = 0; n < 8; ++n) {
    const size_t vbase = ((size_t)t * 8 + n) * 1024 + h * 64;
    const float a = e[n] * inv;
    o0 += a * bf16_to_f32(Vb[vbase + lane]);
    o1 += a * bf16_to_f32(Vb[vbase + lane + 32]);
  }
  AO[qbase + lane]      = f32_to_bf16(o0);
  AO[qbase + lane + 32] = f32_to_bf16(o1);
}

// ---------------------------------------------------------------------------
extern "C" void kernel_launch(void* const* d_in, const int* in_sizes, int n_in,
                              void* d_out, int out_size, void* d_ws, size_t ws_size,
                              hipStream_t stream) {
  (void)in_sizes; (void)n_in; (void)out_size; (void)ws_size;
  const float* current = (const float*)d_in[0];  // (4,2048,1024)
  const float* history = (const float*)d_in[1];  // (4,2048,8,1024)
  const float* Wq      = (const float*)d_in[2];
  const float* Wk      = (const float*)d_in[3];
  const float* Wv      = (const float*)d_in[4];
  const float* Wo      = (const float*)d_in[5];
  float* out = (float*)d_out;

  const int D   = 1024;
  const int Mq  = 4 * 2048;       // 8192
  const int Mkv = 4 * 2048 * 8;   // 65536

  char* ws = (char*)d_ws;
  unsigned short* curB  = (unsigned short*)ws;  ws += (size_t)Mq  * D * 2;
  unsigned short* histB = (unsigned short*)ws;  ws += (size_t)Mkv * D * 2;
  unsigned short* WqB   = (unsigned short*)ws;  ws += (size_t)D * D * 2;
  unsigned short* WkB   = (unsigned short*)ws;  ws += (size_t)D * D * 2;
  unsigned short* WvB   = (unsigned short*)ws;  ws += (size_t)D * D * 2;
  unsigned short* WoB   = (unsigned short*)ws;  ws += (size_t)D * D * 2;
  unsigned short* Qb    = (unsigned short*)ws;  ws += (size_t)Mq  * D * 2;
  unsigned short* Kb    = (unsigned short*)ws;  ws += (size_t)Mkv * D * 2;
  unsigned short* Vb    = (unsigned short*)ws;  ws += (size_t)Mkv * D * 2;
  unsigned short* AO    = (unsigned short*)ws;

  const dim3 blk(256);

  // 0) bulk f32 -> bf16 (grids exact: every size divisible by 2048)
  cvt_bf16_kernel<<<(Mq  * D) / 2048, blk, 0, stream>>>(current, curB);
  cvt_bf16_kernel<<<(Mkv * D) / 2048, blk, 0, stream>>>(history, histB);
  cvt_bf16_kernel<<<(D * D) / 2048, blk, 0, stream>>>(Wq, WqB);
  cvt_bf16_kernel<<<(D * D) / 2048, blk, 0, stream>>>(Wk, WkB);
  cvt_bf16_kernel<<<(D * D) / 2048, blk, 0, stream>>>(Wv, WvB);
  cvt_bf16_kernel<<<(D * D) / 2048, blk, 0, stream>>>(Wo, WoB);

  // 1) Q = cur @ Wq^T
  gemm_wmma_async<1, false><<<dim3(D / 64, Mq / 128), blk, 0, stream>>>(
      curB, WqB, nullptr, Qb, nullptr, nullptr);

  // 2) K,V = hist @ {Wk,Wv}^T (fused over shared history tiles)
  gemm_wmma_async<2, false><<<dim3(D / 64, Mkv / 128), blk, 0, stream>>>(
      histB, WkB, WvB, Kb, Vb, nullptr);

  // 3) softmax over depth
  depth_softmax_kernel<<<(Mq * 16) / 8, blk, 0, stream>>>(Qb, Kb, Vb, AO);

  // 4) out = cur + attn @ Wo^T
  gemm_wmma_async<1, true><<<dim3(D / 64, Mq / 128), blk, 0, stream>>>(
      AO, WoB, nullptr, out, nullptr, current);
}
